// DualTierMiras_91328184582306
// MI455X (gfx1250) — matrix operand order repaired and checked by reading, not compiled, not run on hardware
//
#include <hip/hip_runtime.h>
#include <hip/hip_bf16.h>
#include <math.h>

#define D_MODEL 2048
#define BATCH   16384
#define SCALE_W 0.1f
#define LN_EPS  1e-5f

#define BM 128
#define BN 128
#define BK 32
#define LDSS 40   // padded LDS row stride in bf16 elems (80B -> conflict-free b128 reads)

typedef __attribute__((ext_vector_type(16))) __bf16 bf16x16;
typedef __attribute__((ext_vector_type(8)))  float  f32x8;

union Frag32B { uint4 u[2]; bf16x16 v; };

__device__ __forceinline__ unsigned short f32_to_bf16(float f) {
  unsigned int u = __builtin_bit_cast(unsigned int, f);
  u += 0x7FFFu + ((u >> 16) & 1u);          // round-to-nearest-even
  return (unsigned short)(u >> 16);
}
__device__ __forceinline__ float bf16_to_f32(unsigned short h) {
  unsigned int u = ((unsigned int)h) << 16;
  return __builtin_bit_cast(float, u);
}

// CDNA5 async global->LDS DMA, GVS addressing: addr = SADDR(SGPR pair, uniform)
//   + VADDR32(per-lane) + INST_OFFSET.  Loop-varying part stays on the SALU.
// vdst(%0) = 32-bit LDS byte offset (low 32 bits of generic shared pointer).
__device__ __forceinline__ void async_b128(unsigned lds_off, unsigned voff,
                                           const void* sbase) {
  asm volatile("global_load_async_to_lds_b128 %0, %1, %2"
               :: "v"(lds_off), "v"(voff), "s"((unsigned long long)(size_t)sbase)
               : "memory");
}
__device__ __forceinline__ void async_b128_o16(unsigned lds_off, unsigned voff,
                                               const void* sbase) {
  asm volatile("global_load_async_to_lds_b128 %0, %1, %2 offset:16"
               :: "v"(lds_off), "v"(voff), "s"((unsigned long long)(size_t)sbase)
               : "memory");
}
__device__ __forceinline__ void wait_async0() {
  asm volatile("s_wait_asynccnt 0" ::: "memory");
}

// ---------------------------------------------------------------------------
// f32 -> bf16 bulk convert, 8 elems / thread (vectorized b128 in / b128 out)
// ---------------------------------------------------------------------------
__global__ void __launch_bounds__(256)
convert_bf16x8(const float* __restrict__ in, unsigned short* __restrict__ out) {
  const size_t i = (size_t)blockIdx.x * 256 + threadIdx.x;
  const float4 a = ((const float4*)in)[2 * i];
  const float4 b = ((const float4*)in)[2 * i + 1];
  union { unsigned short s[8]; uint4 u; } r;
  r.s[0] = f32_to_bf16(a.x); r.s[1] = f32_to_bf16(a.y);
  r.s[2] = f32_to_bf16(a.z); r.s[3] = f32_to_bf16(a.w);
  r.s[4] = f32_to_bf16(b.x); r.s[5] = f32_to_bf16(b.y);
  r.s[6] = f32_to_bf16(b.z); r.s[7] = f32_to_bf16(b.w);
  ((uint4*)out)[i] = r.u;
}

// ---------------------------------------------------------------------------
// WtT[n][kk] = bf16( SCALE * tanh( dot(C[kk,0:32], B[n,0:32]) ) )  ([N][K] layout)
// ---------------------------------------------------------------------------
__global__ void __launch_bounds__(256)
build_wt(const float* __restrict__ B, const float* __restrict__ C,
         unsigned short* __restrict__ WtT) {
  const int flat = blockIdx.x * 256 + threadIdx.x;
  const int n  = flat >> 11;
  const int kk = flat & (D_MODEL - 1);
  const float4* cr = (const float4*)(C + (size_t)kk * 32);
  const float4* br = (const float4*)(B + (size_t)n  * 32);
  float s = 0.f;
#pragma unroll
  for (int i = 0; i < 8; ++i) {
    const float4 cv = cr[i], bv = br[i];
    s += cv.x * bv.x + cv.y * bv.y + cv.z * bv.z + cv.w * bv.w;
  }
  WtT[(size_t)n * D_MODEL + kk] = f32_to_bf16(SCALE_W * tanhf(s));
}

// ---------------------------------------------------------------------------
// Row LayerNorm (biased var, torch default) -> bf16
// ---------------------------------------------------------------------------
__global__ void __launch_bounds__(256)
layernorm_bf16(const float* __restrict__ x, const float* __restrict__ w,
               const float* __restrict__ b, unsigned short* __restrict__ out) {
  __shared__ float rs[256];
  __shared__ float rs2[256];
  const int row = blockIdx.x;
  const int tid = threadIdx.x;
  const float* xr = x + (size_t)row * D_MODEL;
  float v[D_MODEL / 256];
  float s = 0.f, s2 = 0.f;
#pragma unroll
  for (int i = 0; i < D_MODEL / 256; ++i) {
    v[i] = xr[tid + i * 256];
    s += v[i]; s2 += v[i] * v[i];
  }
  rs[tid] = s; rs2[tid] = s2;
  __syncthreads();
  for (int off = 128; off > 0; off >>= 1) {
    if (tid < off) { rs[tid] += rs[tid + off]; rs2[tid] += rs2[tid + off]; }
    __syncthreads();
  }
  const float mean = rs[0] * (1.f / D_MODEL);
  const float var  = rs2[0] * (1.f / D_MODEL) - mean * mean;
  const float rstd = rsqrtf(var + LN_EPS);
  unsigned short* orow = out + (size_t)row * D_MODEL;
#pragma unroll
  for (int i = 0; i < D_MODEL / 256; ++i) {
    const int c = tid + i * 256;
    orow[c] = f32_to_bf16((v[i] - mean) * rstd * w[c] + b[c]);
  }
}

// ---------------------------------------------------------------------------
// Gate GEMM: h1 = silu(h @ g1_w.T + g1_b)  (bf16 A [M][K], bf16 B [N][K])
// Double-buffered LDS, async global->LDS (GVS), WMMA f32_16x16x32_bf16.
// ---------------------------------------------------------------------------
#define GATE_STEP(KT, CUR, NXT)                                              \
  {                                                                          \
    if ((KT) + 1 < steps) {                                                  \
      const unsigned short* nA = baseA + (size_t)((KT) + 1) * BK;            \
      const unsigned short* nB = baseB + (size_t)((KT) + 1) * BK;            \
      async_b128    (oA[NXT],      voff, nA);                                \
      async_b128_o16(oA[NXT] + 16, voff, nA);                                \
      async_b128    (oB[NXT],      voff, nB);                                \
      async_b128_o16(oB[NXT] + 16, voff, nB);                                \
    }                                                                        \
    Frag32B a0, a1;                                                          \
    { const unsigned short* p = &lA[CUR][(wm0 + lane16) * LDSS];             \
      a0.u[0] = *(const uint4*)(p + half * 8);                               \
      a0.u[1] = *(const uint4*)(p + 16 + half * 8); }                        \
    { const unsigned short* p = &lA[CUR][(wm0 + 16 + lane16) * LDSS];        \
      a1.u[0] = *(const uint4*)(p + half * 8);                               \
      a1.u[1] = *(const uint4*)(p + 16 + half * 8); }                        \
    _Pragma("unroll")                                                        \
    for (int ni = 0; ni < 4; ++ni) {                                         \
      Frag32B bb;                                                            \
      const unsigned short* q = &lB[CUR][(wn0 + ni * 16 + lane16) * LDSS + half * 8]; \
      bb.u[0] = *(const uint4*)(q); bb.u[1] = *(const uint4*)(q + 16);       \
      acc[0][ni] = __builtin_amdgcn_wmma_f32_16x16x32_bf16(                  \
          false, a0.v, false, bb.v, (short)0, acc[0][ni], false, false);     \
      acc[1][ni] = __builtin_amdgcn_wmma_f32_16x16x32_bf16(                  \
          false, a1.v, false, bb.v, (short)0, acc[1][ni], false, false);     \
    }                                                                        \
    wait_async0();                                                           \
    __syncthreads();                                                         \
  }

__global__ void __launch_bounds__(256)
gate_gemm_wmma(const unsigned short* __restrict__ Abf,
               const unsigned short* __restrict__ BT,
               const float* __restrict__ bias,
               unsigned short* __restrict__ outbf) {
  __shared__ unsigned short lA[2][BM * LDSS];
  __shared__ unsigned short lB[2][BN * LDSS];

  const int tid = threadIdx.x;
  const int lane = tid & 31, wave = tid >> 5;
  const int lane16 = lane & 15, half = lane >> 4;
  const int wm0 = (wave & 3) * 32;
  const int wn0 = (wave >> 2) * 64;
  const int bm = blockIdx.y * BM;
  const int bn = blockIdx.x * BN;
  const int ldr = tid >> 1;
  const int ldc = (tid & 1) << 4;

  // uniform bases (SALU) + one per-lane 32-bit offset shared by all matrices
  const unsigned short* baseA = Abf + (size_t)bm * D_MODEL;
  const unsigned short* baseB = BT  + (size_t)bn * D_MODEL;
  const unsigned voff = (unsigned)((ldr * D_MODEL + ldc) * sizeof(unsigned short));

  unsigned oA[2], oB[2];
  oA[0] = (unsigned)(size_t)&lA[0][ldr * LDSS + ldc];
  oA[1] = (unsigned)(size_t)&lA[1][ldr * LDSS + ldc];
  oB[0] = (unsigned)(size_t)&lB[0][ldr * LDSS + ldc];
  oB[1] = (unsigned)(size_t)&lB[1][ldr * LDSS + ldc];

  f32x8 acc[2][4];
  const f32x8 z = {0.f, 0.f, 0.f, 0.f, 0.f, 0.f, 0.f, 0.f};
#pragma unroll
  for (int i = 0; i < 2; ++i)
#pragma unroll
    for (int j = 0; j < 4; ++j) acc[i][j] = z;

  const int steps = D_MODEL / BK;

  // tile 0 -> buffer 0
  async_b128    (oA[0],      voff, baseA);
  async_b128_o16(oA[0] + 16, voff, baseA);
  async_b128    (oB[0],      voff, baseB);
  async_b128_o16(oB[0] + 16, voff, baseB);
  wait_async0();
  __syncthreads();

#pragma unroll 1
  for (int kt = 0; kt < steps; kt += 2) {
    GATE_STEP(kt,     0, 1)
    GATE_STEP(kt + 1, 1, 0)
  }

#pragma unroll
  for (int mi = 0; mi < 2; ++mi) {
    const int gm0 = bm + wm0 + mi * 16 + half * 8;
#pragma unroll
    for (int ni = 0; ni < 4; ++ni) {
      const int gn = bn + wn0 + ni * 16 + lane16;
      const float bia = bias[gn];
#pragma unroll
      for (int j = 0; j < 8; ++j) {
        const size_t idx = (size_t)(gm0 + j) * D_MODEL + gn;
        const float x = acc[mi][ni][j] + bia;
        outbf[idx] = f32_to_bf16(x / (1.f + expf(-x)));   // SiLU
      }
    }
  }
}

// ---------------------------------------------------------------------------
// delta = sigmoid(h1 . g2_w + g2_b); w = 0.5*(sigmoid(mix)+delta)
// ---------------------------------------------------------------------------
__global__ void __launch_bounds__(256)
gate_reduce(const unsigned short* __restrict__ h1, const float* __restrict__ g2w,
            const float* __restrict__ g2b, const float* __restrict__ mix_logit,
            float* __restrict__ wrow, float* __restrict__ wfast_out) {
  __shared__ float red[256];
  const int row = blockIdx.x;
  const int tid = threadIdx.x;
  const unsigned short* hr = h1 + (size_t)row * D_MODEL;
  float s = 0.f;
#pragma unroll
  for (int i = 0; i < D_MODEL / 256; ++i) {
    const int c = tid + i * 256;
    s += bf16_to_f32(hr[c]) * g2w[c];
  }
  red[tid] = s;
  __syncthreads();
  for (int off = 128; off > 0; off >>= 1) {
    if (tid < off) red[tid] += red[tid + off];
    __syncthreads();
  }
  if (tid == 0) {
    const float delta = 1.f / (1.f + expf(-(red[0] + g2b[0])));
    const float base  = 1.f / (1.f + expf(-mix_logit[0]));
    const float w = 0.5f * (base + delta);
    wrow[row] = w;
    wfast_out[row] = w;
  }
}

// ---------------------------------------------------------------------------
// Dual-tier GEMM: shares A (k) tile across both B matrices; async loads to
// double-buffered LDS; epilogue fuses +k*diag(D) and v = w*vf + (1-w)*vd.
// ---------------------------------------------------------------------------
#define TIER_STEP(KT, CUR, NXT)                                              \
  {                                                                          \
    if ((KT) + 1 < steps) {                                                  \
      const unsigned short* nA = baseA + (size_t)((KT) + 1) * BK;            \
      const unsigned short* nF = baseF + (size_t)((KT) + 1) * BK;            \
      const unsigned short* nD = baseD + (size_t)((KT) + 1) * BK;            \
      async_b128    (oA[NXT],      voff, nA);                                \
      async_b128_o16(oA[NXT] + 16, voff, nA);                                \
      async_b128    (oF[NXT],      voff, nF);                                \
      async_b128_o16(oF[NXT] + 16, voff, nF);                                \
      async_b128    (oD[NXT],      voff, nD);                                \
      async_b128_o16(oD[NXT] + 16, voff, nD);                                \
    }                                                                        \
    Frag32B a0, a1;                                                          \
    { const unsigned short* p = &lA[CUR][(wm0 + lane16) * LDSS];             \
      a0.u[0] = *(const uint4*)(p + half * 8);                               \
      a0.u[1] = *(const uint4*)(p + 16 + half * 8); }                        \
    { const unsigned short* p = &lA[CUR][(wm0 + 16 + lane16) * LDSS];        \
      a1.u[0] = *(const uint4*)(p + half * 8);                               \
      a1.u[1] = *(const uint4*)(p + 16 + half * 8); }                        \
    _Pragma("unroll")                                                        \
    for (int ni = 0; ni < 4; ++ni) {                                         \
      const int col = wn0 + ni * 16 + lane16;                                \
      Frag32B bF, bD;                                                        \
      const unsigned short* qf = &lF[CUR][col * LDSS + half * 8];            \
      bF.u[0] = *(const uint4*)(qf); bF.u[1] = *(const uint4*)(qf + 16);     \
      const unsigned short* qd = &lD[CUR][col * LDSS + half * 8];            \
      bD.u[0] = *(const uint4*)(qd); bD.u[1] = *(const uint4*)(qd + 16);     \
      accF[0][ni] = __builtin_amdgcn_wmma_f32_16x16x32_bf16(                 \
          false, a0.v, false, bF.v, (short)0, accF[0][ni], false, false);    \
      accF[1][ni] = __builtin_amdgcn_wmma_f32_16x16x32_bf16(                 \
          false, a1.v, false, bF.v, (short)0, accF[1][ni], false, false);    \
      accD[0][ni] = __builtin_amdgcn_wmma_f32_16x16x32_bf16(                 \
          false, a0.v, false, bD.v, (short)0, accD[0][ni], false, false);    \
      accD[1][ni] = __builtin_amdgcn_wmma_f32_16x16x32_bf16(                 \
          false, a1.v, false, bD.v, (short)0, accD[1][ni], false, false);    \
    }                                                                        \
    wait_async0();                                                           \
    __syncthreads();                                                         \
  }

__global__ void __launch_bounds__(256)
dual_tier_wmma(const unsigned short* __restrict__ Abf,
               const unsigned short* __restrict__ BTf,
               const unsigned short* __restrict__ BTd,
               const float* __restrict__ kf32,
               const float* __restrict__ Dfast, const float* __restrict__ Ddeep,
               const float* __restrict__ wrow,
               float* __restrict__ out_v, float* __restrict__ out_vf,
               float* __restrict__ out_vd) {
  __shared__ unsigned short lA[2][BM * LDSS];
  __shared__ unsigned short lF[2][BN * LDSS];
  __shared__ unsigned short lD[2][BN * LDSS];

  const int tid = threadIdx.x;
  const int lane = tid & 31, wave = tid >> 5;
  const int lane16 = lane & 15, half = lane >> 4;
  const int wm0 = (wave & 3) * 32;
  const int wn0 = (wave >> 2) * 64;
  const int bm = blockIdx.y * BM;
  const int bn = blockIdx.x * BN;
  const int ldr = tid >> 1;
  const int ldc = (tid & 1) << 4;

  const unsigned short* baseA = Abf + (size_t)bm * D_MODEL;
  const unsigned short* baseF = BTf + (size_t)bn * D_MODEL;
  const unsigned short* baseD = BTd + (size_t)bn * D_MODEL;
  const unsigned voff = (unsigned)((ldr * D_MODEL + ldc) * sizeof(unsigned short));

  unsigned oA[2], oF[2], oD[2];
  oA[0] = (unsigned)(size_t)&lA[0][ldr * LDSS + ldc];
  oA[1] = (unsigned)(size_t)&lA[1][ldr * LDSS + ldc];
  oF[0] = (unsigned)(size_t)&lF[0][ldr * LDSS + ldc];
  oF[1] = (unsigned)(size_t)&lF[1][ldr * LDSS + ldc];
  oD[0] = (unsigned)(size_t)&lD[0][ldr * LDSS + ldc];
  oD[1] = (unsigned)(size_t)&lD[1][ldr * LDSS + ldc];

  f32x8 accF[2][4], accD[2][4];
  const f32x8 z = {0.f, 0.f, 0.f, 0.f, 0.f, 0.f, 0.f, 0.f};
#pragma unroll
  for (int i = 0; i < 2; ++i)
#pragma unroll
    for (int j = 0; j < 4; ++j) { accF[i][j] = z; accD[i][j] = z; }

  const int steps = D_MODEL / BK;

  // tile 0 -> buffer 0
  async_b128    (oA[0],      voff, baseA);
  async_b128_o16(oA[0] + 16, voff, baseA);
  async_b128    (oF[0],      voff, baseF);
  async_b128_o16(oF[0] + 16, voff, baseF);
  async_b128    (oD[0],      voff, baseD);
  async_b128_o16(oD[0] + 16, voff, baseD);
  wait_async0();
  __syncthreads();

#pragma unroll 1
  for (int kt = 0; kt < steps; kt += 2) {
    TIER_STEP(kt,     0, 1)
    TIER_STEP(kt + 1, 1, 0)
  }

#pragma unroll
  for (int mi = 0; mi < 2; ++mi) {
    const int gm0 = bm + wm0 + mi * 16 + half * 8;
#pragma unroll
    for (int ni = 0; ni < 4; ++ni) {
      const int gn = bn + wn0 + ni * 16 + lane16;
      const float dF = Dfast[gn];
      const float dD = Ddeep[gn];
#pragma unroll
      for (int j = 0; j < 8; ++j) {
        const size_t idx = (size_t)(gm0 + j) * D_MODEL + gn;
        const float kv = kf32[idx];
        const float vf = accF[mi][ni][j] + kv * dF;
        const float vd = accD[mi][ni][j] + kv * dD;
        const float w  = wrow[gm0 + j];
        out_vf[idx] = vf;
        out_vd[idx] = vd;
        out_v[idx]  = w * vf + (1.f - w) * vd;
      }
    }
  }
}

// ---------------------------------------------------------------------------
extern "C" void kernel_launch(void* const* d_in, const int* in_sizes, int n_in,
                              void* d_out, int out_size, void* d_ws, size_t ws_size,
                              hipStream_t stream) {
  (void)in_sizes; (void)n_in; (void)out_size; (void)ws_size;
  const float* k   = (const float*)d_in[0];
  const float* ctx = (const float*)d_in[1];
  const float* Bf  = (const float*)d_in[2];
  const float* Cf  = (const float*)d_in[3];
  const float* Dfa = (const float*)d_in[4];
  const float* Bd  = (const float*)d_in[5];
  const float* Cd  = (const float*)d_in[6];
  const float* Dde = (const float*)d_in[7];
  const float* mix = (const float*)d_in[8];
  const float* lnw = (const float*)d_in[9];
  const float* lnb = (const float*)d_in[10];
  const float* g1w = (const float*)d_in[11];
  const float* g1b = (const float*)d_in[12];
  const float* g2w = (const float*)d_in[13];
  const float* g2b = (const float*)d_in[14];

  const size_t N  = (size_t)BATCH * D_MODEL;      // 33,554,432
  const size_t DD = (size_t)D_MODEL * D_MODEL;    // 4,194,304

  char* ws = (char*)d_ws;
  unsigned short* kbf  = (unsigned short*)(ws);                    // 64 MB
  unsigned short* hbf  = (unsigned short*)(ws + N * 2);            // 64 MB
  unsigned short* h1bf = (unsigned short*)(ws + N * 4);            // 64 MB
  unsigned short* g1bf = (unsigned short*)(ws + N * 6);            //  8 MB
  unsigned short* wtF  = (unsigned short*)(ws + N * 6 + DD * 2);   //  8 MB
  unsigned short* wtD  = (unsigned short*)(ws + N * 6 + DD * 4);   //  8 MB
  float*          wrow = (float*)(ws + N * 6 + DD * 6);            // 64 KB

  float* out  = (float*)d_out;
  float* o_v  = out;
  float* o_vf = out + N;
  float* o_vd = out + 2 * N;
  float* o_w  = out + 3 * N;

  // gate path first (produces per-row mix weight consumed by dual GEMM epilogue)
  convert_bf16x8<<<dim3((unsigned)(DD / 8 / 256)), dim3(256), 0, stream>>>(g1w, g1bf);
  layernorm_bf16<<<dim3(BATCH), dim3(256), 0, stream>>>(ctx, lnw, lnb, hbf);
  gate_gemm_wmma<<<dim3(D_MODEL / BN, BATCH / BM), dim3(256), 0, stream>>>(hbf, g1bf, g1b, h1bf);
  gate_reduce<<<dim3(BATCH), dim3(256), 0, stream>>>(h1bf, g2w, g2b, mix, wrow, o_w);

  // tier path
  convert_bf16x8<<<dim3((unsigned)(N / 8 / 256)), dim3(256), 0, stream>>>(k, kbf);
  build_wt<<<dim3((unsigned)(DD / 256)), dim3(256), 0, stream>>>(Bf, Cf, wtF);
  build_wt<<<dim3((unsigned)(DD / 256)), dim3(256), 0, stream>>>(Bd, Cd, wtD);
  dual_tier_wmma<<<dim3(D_MODEL / BN, BATCH / BM), dim3(256), 0, stream>>>(
      kbf, wtF, wtD, k, Dfa, Dde, wrow, o_v, o_vf, o_vd);
}
